// GCNConv_86311662780630
// MI455X (gfx1250) — matrix operand order repaired and checked by reading, not compiled, not run on hardware
//
#include <hip/hip_runtime.h>

typedef __attribute__((ext_vector_type(16))) __bf16 v16bf;
typedef __attribute__((ext_vector_type(8)))  float  v8f;
typedef __attribute__((ext_vector_type(4)))  int    v4i;

#define N_NODES 50000
#define D_FEAT  512
#define UNITS   512

// CDNA5 async global->LDS path (guarded: present on this toolchain per round-2
// diagnostics; falls back to VGPR round-trip copies if the builtins are missing)
#if defined(__AMDGCN__) && __has_builtin(__builtin_amdgcn_global_load_async_to_lds_b128) && \
    __has_builtin(__builtin_amdgcn_s_wait_asynccnt)
#define USE_ASYNC_LDS 1
typedef __attribute__((address_space(1))) v4i* g_v4i_p;   // global int4*
typedef __attribute__((address_space(3))) v4i* l_v4i_p;   // LDS int4*
#else
#define USE_ASYNC_LDS 0
#endif

// ---------- helpers ----------
__device__ __forceinline__ unsigned short f2bf(float f) {
  unsigned u = __builtin_bit_cast(unsigned, f);
  unsigned r = u + 0x7FFFu + ((u >> 16) & 1u);   // round-to-nearest-even
  return (unsigned short)(r >> 16);
}

// ---------- kernel 1: W [K][N] fp32 -> W^T [N][K] bf16 ----------
__global__ void convertW_T(const float* __restrict__ W, unsigned short* __restrict__ WT) {
  int i = blockIdx.x * blockDim.x + threadIdx.x;   // over 512*512
  int n = i >> 9;
  int k = i & 511;
  WT[n * D_FEAT + k] = f2bf(W[k * UNITS + n]);
}

// ---------- kernel 2: out[n][u] = bias[u] (accumulator init) ----------
__global__ void init_bias(const float* __restrict__ bias, float* __restrict__ out) {
  int i = blockIdx.x * blockDim.x + threadIdx.x;   // over N*U/4
  float4 b = ((const float4*)bias)[i & 127];
  ((float4*)out)[i] = b;
}

// ---------- kernel 3: h = X @ W via bf16 WMMA, fp32 accumulate ----------
// Block tile: M=64, N=128. 256 threads = 8 waves in a 4(M) x 2(N) grid.
// Each wave: 16(M) x 64(N) via 4 accumulators of 16x16 f32.
__global__ void __launch_bounds__(256, 2)
gemm_wmma_bf16(const float* __restrict__ X, const unsigned short* __restrict__ WT,
               float* __restrict__ h) {
  // rows padded to 80B (40 bf16) to spread LDS banks; 16B aligned (80 = 5*16)
  __shared__ __align__(16) unsigned short Xs[64 * 40];
  __shared__ __align__(16) unsigned short Ws[128 * 40];

  const int tid  = threadIdx.x;
  const int lane = tid & 31;
  const int wave = tid >> 5;
  const int wm   = wave >> 1;     // 0..3  (M sub-tile)
  const int wn   = wave & 1;      // 0..1  (N sub-tile)
  const int half = lane >> 4;     // lane half selects K sub-range per ISA layout
  const int l16  = lane & 15;

  const int m0 = blockIdx.y * 64;
  const int n0 = blockIdx.x * 128;

  v8f acc[4] = {};

  // staging roles
  const int xr  = tid >> 2;       // 0..63   Xs row
  const int xs  = tid & 3;        // 16B segment (8 floats)
  const int wr  = tid >> 1;       // 0..127  Ws row
  const int wsg = tid & 1;        // 32B segment (16 bf16)

  for (int k0 = 0; k0 < D_FEAT; k0 += 32) {
    // ---- stage W^T tile (already bf16; row = N, cols = K) ----
    // Async DMA global->LDS: no VGPR round trip, tracked by ASYNCcnt.
    {
      const unsigned short* gsrc = WT + (size_t)(n0 + wr) * D_FEAT + k0 + wsg * 16;
      char* ldst = (char*)(Ws + wr * 40) + wsg * 32;
#if USE_ASYNC_LDS
      __builtin_amdgcn_global_load_async_to_lds_b128(
          (g_v4i_p)(const void*)gsrc, (l_v4i_p)(void*)ldst, 0, 0);
      __builtin_amdgcn_global_load_async_to_lds_b128(
          (g_v4i_p)(const void*)(gsrc + 8), (l_v4i_p)(void*)(ldst + 16), 0, 0);
#else
      const uint4* p = (const uint4*)gsrc;
      uint4 q0 = p[0];
      uint4 q1 = p[1];
      uint4* d = (uint4*)ldst;
      d[0] = q0;
      d[1] = q1;
#endif
    }
    // ---- stage X tile, converting fp32 -> bf16 (overlaps async W DMA) ----
    {
      int gm = m0 + xr;
      float4 f0 = {}, f1 = {};
      if (gm < N_NODES) {
        const float4* p = (const float4*)(X + (size_t)gm * D_FEAT + k0 + xs * 8);
        f0 = p[0];
        f1 = p[1];
      }
      uint4 q;
      q.x = (unsigned)f2bf(f0.x) | ((unsigned)f2bf(f0.y) << 16);
      q.y = (unsigned)f2bf(f0.z) | ((unsigned)f2bf(f0.w) << 16);
      q.z = (unsigned)f2bf(f1.x) | ((unsigned)f2bf(f1.y) << 16);
      q.w = (unsigned)f2bf(f1.z) | ((unsigned)f2bf(f1.w) << 16);
      *(uint4*)((char*)(Xs + xr * 40) + xs * 16) = q;
    }
#if USE_ASYNC_LDS
    __builtin_amdgcn_s_wait_asynccnt(0);
#endif
    __syncthreads();

    // ---- fragment loads (two ds_load_b128 each) + WMMA ----
    union Frag { uint4 q[2]; v16bf v; } a, b;
    {
      // 16-bit A 16x32 layout: lanes0-15 K=0..7 & 16..23, lanes16-31 K=8..15 & 24..31
      const char* rowp = (const char*)(Xs + (wm * 16 + l16) * 40);
      a.q[0] = *(const uint4*)(rowp + half * 16);
      a.q[1] = *(const uint4*)(rowp + 32 + half * 16);
    }
#pragma unroll
    for (int j = 0; j < 4; ++j) {
      // 16-bit B 32x16 layout (matches A): lanes0-15 K=0..7 & 16..23, lanes16-31 K=8..15 & 24..31
      const char* rowp = (const char*)(Ws + (wn * 64 + j * 16 + l16) * 40);
      b.q[0] = *(const uint4*)(rowp + half * 16);
      b.q[1] = *(const uint4*)(rowp + 32 + half * 16);
      acc[j] = __builtin_amdgcn_wmma_f32_16x16x32_bf16(
          /*neg_a=*/false, a.v, /*neg_b=*/false, b.v,
          /*c_mod=*/(short)0, acc[j], /*reuse_a=*/false, /*reuse_b=*/false);
    }
    __syncthreads();
  }

  // ---- epilogue: C layout -> h (fp32). VGPR r: lanes0-15 M=r, lanes16-31 M=8+r
#pragma unroll
  for (int j = 0; j < 4; ++j) {
    int n = n0 + wn * 64 + j * 16 + l16;
#pragma unroll
    for (int r = 0; r < 8; ++r) {
      int m = m0 + wm * 16 + half * 8 + r;
      if (m < N_NODES) h[(size_t)m * UNITS + n] = acc[j][r];
    }
  }
}

// ---------- kernel 4: edge-parallel SpMM, atomics into out (L2-resident) ----------
__global__ void spmm_atomic(const float* __restrict__ h, const float* __restrict__ ew,
                            const int* __restrict__ esrc, const int* __restrict__ edst,
                            float* __restrict__ out, int E, int totalWaves) {
  const int wave = (blockIdx.x * blockDim.x + threadIdx.x) >> 5;
  const int lane = threadIdx.x & 31;
  for (int e = wave; e < E; e += totalWaves) {
    const int s = esrc[e];
    const int d = edst[e];
    const float w = ew[e];
    // prefetch next iteration's source row into cache hierarchy
    const int en = e + totalWaves;
    if (en < E) {
      int sn = esrc[en];
      __builtin_prefetch(&h[(size_t)sn * UNITS + lane * 16], 0, 0);
    }
    const float4* hr = (const float4*)(h + (size_t)s * UNITS + lane * 16);
    float* orow = out + (size_t)d * UNITS + lane * 16;
#pragma unroll
    for (int c = 0; c < 4; ++c) {
      float4 v = hr[c];
      atomicAdd(orow + c * 4 + 0, v.x * w);
      atomicAdd(orow + c * 4 + 1, v.y * w);
      atomicAdd(orow + c * 4 + 2, v.z * w);
      atomicAdd(orow + c * 4 + 3, v.w * w);
    }
  }
}

// ---------- kernel 5: in-place ReLU ----------
__global__ void relu_inplace(float* __restrict__ out) {
  int i = blockIdx.x * blockDim.x + threadIdx.x;   // over N*U/4
  float4 v = ((float4*)out)[i];
  v.x = fmaxf(v.x, 0.f);
  v.y = fmaxf(v.y, 0.f);
  v.z = fmaxf(v.z, 0.f);
  v.w = fmaxf(v.w, 0.f);
  ((float4*)out)[i] = v;
}

// ---------- launcher ----------
extern "C" void kernel_launch(void* const* d_in, const int* in_sizes, int n_in,
                              void* d_out, int out_size, void* d_ws, size_t ws_size,
                              hipStream_t stream) {
  const float* X    = (const float*)d_in[0];
  const float* W    = (const float*)d_in[1];
  const float* bias = (const float*)d_in[2];
  const float* ew   = (const float*)d_in[3];
  const int*   esrc = (const int*)d_in[4];
  const int*   edst = (const int*)d_in[5];
  float* out = (float*)d_out;
  const int E = in_sizes[3];

  // workspace: h [N_NODES x UNITS] fp32, then W^T bf16
  float* h = (float*)d_ws;
  unsigned short* WT =
      (unsigned short*)((char*)d_ws + (size_t)N_NODES * UNITS * sizeof(float));

  convertW_T<<<(D_FEAT * UNITS) / 256, 256, 0, stream>>>(W, WT);

  const int vec4 = (N_NODES * UNITS) / 4;           // 6,400,000
  init_bias<<<vec4 / 256, 256, 0, stream>>>(bias, out);

  dim3 g(UNITS / 128, (N_NODES + 63) / 64);          // (4, 782)
  gemm_wmma_bf16<<<g, 256, 0, stream>>>(X, WT, h);

  const int blocks = 2048;                           // 16384 waves, grid-stride
  spmm_atomic<<<blocks, 256, 0, stream>>>(h, ew, esrc, edst, out, E, blocks * 8);

  relu_inplace<<<vec4 / 256, 256, 0, stream>>>(out);
}
